// Block_Cross_5600637354006
// MI455X (gfx1250) — matrix operand order repaired and checked by reading, not compiled, not run on hardware
//
#include <hip/hip_runtime.h>
#include <hip/hip_bf16.h>

// ---------------------------------------------------------------------------
// CDNA5 (gfx1250) bf16 WMMA implementation of the low-rank attention block.
// Heavy GEMMs: v_wmma_f32_16x16x32_bf16 (wave32). bf16 operand staging uses
// the Tensor Data Mover (tensor_load_to_lds) with LDS padding + double buffer.
// ---------------------------------------------------------------------------

typedef __attribute__((ext_vector_type(16))) __bf16 v16bf;
typedef __attribute__((ext_vector_type(8)))  float  v8f;
typedef __attribute__((ext_vector_type(4)))  unsigned int u32x4;
typedef __attribute__((ext_vector_type(8)))  int  i32x8;
typedef __attribute__((ext_vector_type(4)))  int  i32x4;

struct Frag32B { uint4 a, b; };

// Native f32 -> bf16 (RNE) via hardware convert (v_cvt_pk_bf16_f32)
static __device__ __forceinline__ unsigned short f2bf(float f) {
  return __builtin_bit_cast(unsigned short, (__bf16)f);
}

// A-fragment (16x32 bf16): lane m = lane%16, halves split K as {k0..k0+7, k0+16..k0+23}
static __device__ __forceinline__ v16bf frag_a(const unsigned short* p, int k0) {
  Frag32B f;
  f.a = *(const uint4*)(p + k0);
  f.b = *(const uint4*)(p + k0 + 16);
  return __builtin_bit_cast(v16bf, f);
}
// B-fragment from Bt[n][k] row-major: lane n = lane%16 reads k = {kb/2..kb/2+7, kb/2+16..kb/2+23}
static __device__ __forceinline__ v16bf frag_b(const unsigned short* p, int kb) {
  Frag32B f;
  f.a = *(const uint4*)(p + (kb >> 1));
  f.b = *(const uint4*)(p + (kb >> 1) + 16);
  return __builtin_bit_cast(v16bf, f);
}

static __device__ __forceinline__ v8f zero8() {
  v8f z = {0.f, 0.f, 0.f, 0.f, 0.f, 0.f, 0.f, 0.f};
  return z;
}

static __device__ __forceinline__ v8f bwmma(v16bf a, v16bf b, v8f c) {
  return __builtin_amdgcn_wmma_f32_16x16x32_bf16(false, a, false, b, (short)0, c,
                                                 false, false);
}

// 16-element global->bf16 converters
static __device__ __forceinline__ void cvt16(unsigned short* dst, const float* src) {
#pragma unroll
  for (int i = 0; i < 16; i += 4) {
    float4 f = *(const float4*)(src + i);
    dst[i + 0] = f2bf(f.x); dst[i + 1] = f2bf(f.y);
    dst[i + 2] = f2bf(f.z); dst[i + 3] = f2bf(f.w);
  }
}
static __device__ __forceinline__ void cvt16(unsigned short* dst, const unsigned short* src) {
  *(uint4*)(dst)     = *(const uint4*)(src);
  *(uint4*)(dst + 8) = *(const uint4*)(src + 8);
}

#if __has_builtin(__builtin_amdgcn_tensor_load_to_lds)
#define HAVE_TDM 1

static __device__ __forceinline__ void tdm_issue(u32x4 g0, i32x8 g1) {
  i32x4 z4 = {0, 0, 0, 0};
#if defined(__clang_major__) && __clang_major__ >= 23
  i32x8 z8 = {0, 0, 0, 0, 0, 0, 0, 0};
  __builtin_amdgcn_tensor_load_to_lds(g0, g1, z4, z4, z8, 0);
#else
  __builtin_amdgcn_tensor_load_to_lds(g0, g1, z4, z4, 0);
#endif
}

static __device__ __forceinline__ u32x4 tdm_g0(const void* gptr, void* ldsptr) {
  const unsigned long long ga = (unsigned long long)(size_t)gptr;
  // count=1 | lds_addr | global_addr[56:0] | type=2
  u32x4 g0 = { 1u,
               (unsigned)(size_t)ldsptr,     // generic LDS addr[31:0] = LDS offset
               (unsigned)ga,
               (unsigned)((ga >> 32) & 0x01FFFFFFu) | (2u << 30) };
  return g0;
}

// TDM: 16 rows x 2048B contiguous block; pad 16B after every 128B
// (pad_interval=4 -> 32 DWORDs, pad_amount=3 -> 4 DWORDs). LDS result:
// rows of 64 bf16 at a 72-halfword (144B) stride == Mbuf[h*72].
static __device__ __forceinline__ void tdm_load_kv(const void* gptr, void* ldsptr) {
  i32x8 g1 = { (int)((3u << 16) | (1u << 20) | (4u << 22) | (3u << 25)),
               (int)(256u << 16),          // tensor_dim0[15:0]<<16
               (int)(16u << 16),           // tensor_dim0[31:16] | tensor_dim1[15:0]<<16
               (int)(256u << 16),          // tensor_dim1[31:16] | tile_dim0<<16
               16,                         // tile_dim1 | tile_dim2<<16
               256,                        // tensor_dim0_stride[31:0]  (8B units)
               0, 0 };
  tdm_issue(tdm_g0(gptr, ldsptr), g1);
}

// TDM: GEMM A tile, 128 rows x 64B, row stride Kdim bf16 elements; pad 16B
// after every 64B (pad_interval=3, pad_amount=3) -> 40-halfword (80B) stride.
static __device__ __forceinline__ void tdm_load_gemmA(const void* gptr, void* ldsptr,
                                                      int Kdim) {
  i32x8 g1 = { (int)((3u << 16) | (1u << 20) | (3u << 22) | (3u << 25)),
               (int)(8u << 16),            // tensor_dim0 = 8 (8B units)
               (int)(128u << 16),          // tensor_dim1 = 128
               (int)(8u << 16),            // tile_dim0 = 8
               128,                        // tile_dim1 = 128
               Kdim >> 2,                  // tensor_dim0_stride (8B units)
               0, 0 };
  tdm_issue(tdm_g0(gptr, ldsptr), g1);
}
#else
#define HAVE_TDM 0
#endif

// ---------------------------------------------------------------------------
// Generic GEMM:  C[M,N] = act(A[M,K] @ W[N,K]^T + bias[N])
// A is f32 or bf16(u16), W is f32, C is f32 or bf16. Tile 128x128x32, 8 waves.
// bf16 A tiles are staged by the TDM (double buffered); f32 A and f32 W tiles
// are loaded + converted by the VALU (v_cvt_pk_bf16_f32).
// ---------------------------------------------------------------------------
template <typename TA, bool OUT_BF16, bool RELU>
__global__ __launch_bounds__(256) void gemm_bias_kernel(
    const TA* __restrict__ A, const float* __restrict__ W,
    const float* __restrict__ bias, void* __restrict__ Cout,
    int M, int N, int K) {
  constexpr bool A_TDM = (sizeof(TA) == 2) && (HAVE_TDM != 0);

  __shared__ unsigned short As[2][128 * 40];
  __shared__ unsigned short Bs[128 * 40];

  const int tid  = threadIdx.x;
  const int wave = tid >> 5;
  const int lane = tid & 31;
  const int fr   = lane & 15;
  const int hi   = lane >> 4;
  const int k0   = hi << 3;   // A-frag K split
  const int kb   = hi << 4;   // B-frag K split
  const int dm   = hi << 3;   // D-tile row offset

  const int  wr = wave >> 2;              // wave M position (0..1) * 64
  const int  wc = wave & 3;               // wave N position (0..3) * 32
  const long m0 = (long)blockIdx.y * 128;
  const long n0 = (long)blockIdx.x * 128;

  const int lr = tid >> 1;                // tile row this thread loads
  const int lc = (tid & 1) << 4;          // 0 or 16

  const TA*    Abase = A + m0 * (long)K;               // TDM path
  const TA*    Ag = A + (m0 + lr) * (long)K + lc;      // manual path
  const float* Wg = W + (n0 + lr) * (long)K + lc;

  v8f acc[4][2];
#pragma unroll
  for (int i = 0; i < 4; ++i)
#pragma unroll
    for (int j = 0; j < 2; ++j) acc[i][j] = zero8();

#if HAVE_TDM
  if constexpr (A_TDM) {
    if (wave == 0) tdm_load_gemmA(Abase, &As[0][0], K);
  }
#endif

  int p = 0;
  for (int kc = 0; kc < K; kc += 32) {
    unsigned short tB[16];
    cvt16(tB, Wg + kc);
    __builtin_prefetch(Wg + kc + 64, 0, 0);

    if constexpr (A_TDM) {
#if HAVE_TDM
      if (wave == 0 && kc + 32 < K)
        tdm_load_gemmA(Abase + kc + 32, &As[p ^ 1][0], K);
#endif
    } else {
      unsigned short tA[16];
      cvt16(tA, Ag + kc);
      __builtin_prefetch(Ag + kc + 64, 0, 0);
      *(uint4*)&As[0][lr * 40 + lc]     = *(const uint4*)&tA[0];
      *(uint4*)&As[0][lr * 40 + lc + 8] = *(const uint4*)&tA[8];
    }
    *(uint4*)&Bs[lr * 40 + lc]     = *(const uint4*)&tB[0];
    *(uint4*)&Bs[lr * 40 + lc + 8] = *(const uint4*)&tB[8];

#if HAVE_TDM
    if constexpr (A_TDM) {
      if (wave == 0) {
        if (kc + 32 < K) __builtin_amdgcn_s_wait_tensorcnt(1);
        else             __builtin_amdgcn_s_wait_tensorcnt(0);
      }
    }
#endif
    __syncthreads();

    const unsigned short* Acur = A_TDM ? &As[p][0] : &As[0][0];
    v16bf bfr[2];
#pragma unroll
    for (int nt = 0; nt < 2; ++nt)
      bfr[nt] = frag_b(&Bs[(wc * 32 + nt * 16 + fr) * 40], kb);
#pragma unroll
    for (int mt = 0; mt < 4; ++mt) {
      v16bf af = frag_a(&Acur[(wr * 64 + mt * 16 + fr) * 40], k0);
#pragma unroll
      for (int nt = 0; nt < 2; ++nt)
        acc[mt][nt] = bwmma(af, bfr[nt], acc[mt][nt]);
    }
    __syncthreads();
    p ^= 1;
  }

  // Epilogue: bias (+ReLU), store f32 or bf16
#pragma unroll
  for (int mt = 0; mt < 4; ++mt) {
    const long mbase = m0 + wr * 64 + mt * 16 + dm;
#pragma unroll
    for (int nt = 0; nt < 2; ++nt) {
      const long  n  = n0 + wc * 32 + nt * 16 + fr;
      const float bv = bias[n];
#pragma unroll
      for (int i = 0; i < 8; ++i) {
        float v = acc[mt][nt][i] + bv;
        if (RELU) v = v > 0.f ? v : 0.f;
        const size_t off = (size_t)(mbase + i) * (size_t)N + (size_t)n;
        if (OUT_BF16) ((unsigned short*)Cout)[off] = f2bf(v);
        else          ((float*)Cout)[off] = v;
      }
    }
  }
}

// ---------------------------------------------------------------------------
// Fused low-rank attention middle. One block per (b, a); 256 threads, 8 waves.
//   q[h][l]  = Qp[b, 32a + h/8,  (h%8)*128 + l]     h<256, l<128
//   km[h][kk]= Kp[b, 16a + h/16, (h%16)*64 + kk]    kk<64  (same for V)
//   k_lr = km @ fac[a]   (256x16),  v_lr likewise
//   scores[l][r] = (1/16) sum_h q[h][l] k_lr[h][r];  softmax over r (16)
//   ctx[l][h] = sum_r attn[l][r] v_lr[h][r]
//   ctx -> global [b, 32a + l/4, (l%4)*256 + h]  (bf16)
// K and V tiles are both DMA'd up front by the TDM; in-order per-wave
// completion lets the k_lr GEMM overlap the V transfer.
// ---------------------------------------------------------------------------
__global__ __launch_bounds__(256) void attn_kernel(
    const unsigned short* __restrict__ Qp,   // [16,2048,1024] bf16
    const unsigned short* __restrict__ Kp,   // [16,1024,1024] bf16
    const unsigned short* __restrict__ Vp,   // [16,1024,1024] bf16
    const float* __restrict__ factor,        // [64,64,16]
    unsigned short* __restrict__ ctx)        // [16,2048,1024] bf16
{
  __shared__ unsigned short Kbuf[256 * 72];  // Kmat            (36 KB)
  __shared__ unsigned short Vbuf[256 * 72];  // Vmat            (36 KB)
  __shared__ unsigned short facT[16 * 72];   // fac^T [r][kk]   (2.25 KB)
  __shared__ unsigned short klrT[16 * 264];  // k_lr^T [r][h]   (8.25 KB)
  __shared__ unsigned short vlrB[256 * 40];  // v_lr [h][r,pad] (20 KB)
  __shared__ unsigned short qT[128 * 40];    // q^T chunk       (10 KB)
  __shared__ float          scoresS[128 * 16];
  __shared__ unsigned short attnB[128 * 40];

  const int tid  = threadIdx.x;
  const int wave = tid >> 5;
  const int lane = tid & 31;
  const int fr   = lane & 15;
  const int hi   = lane >> 4;
  const int k0   = hi << 3;
  const int kb   = hi << 4;
  const int dm   = hi << 3;

  const int b = blockIdx.x >> 6;
  const int a = blockIdx.x & 63;

  const unsigned short* kbase = Kp + (size_t)(b * 1024 + a * 16) * 1024;
  const unsigned short* vbase = Vp + (size_t)(b * 1024 + a * 16) * 1024;

#if HAVE_TDM
  if (wave == 0) {                 // issue both DMAs immediately
    tdm_load_kv(kbase, &Kbuf[0]);
    tdm_load_kv(vbase, &Vbuf[0]);
  }
#else
  {
    const int row = tid >> 4, col = (tid & 15) * 64;
#pragma unroll
    for (int i = 0; i < 64; i += 8) {
      *(uint4*)&Kbuf[tid * 72 + i] = *(const uint4*)(kbase + row * 1024 + col + i);
      *(uint4*)&Vbuf[tid * 72 + i] = *(const uint4*)(vbase + row * 1024 + col + i);
    }
  }
#endif

  // --- factor^T and v_lr zero-pad (overlaps the DMAs) ----------------------
  const float* fa = factor + a * 1024;       // [kk][r] row-major
  for (int i = tid; i < 1024; i += 256) {
    const int kk = i >> 4, r = i & 15;
    facT[r * 72 + kk] = f2bf(fa[i]);
  }
  for (int i = tid; i < 4096; i += 256) {    // vlrB columns 16..31 = 0
    const int h = i >> 4, r = i & 15;
    vlrB[h * 40 + 16 + r] = 0;
  }

#if HAVE_TDM
  if (wave == 0) __builtin_amdgcn_s_wait_tensorcnt(1);  // K tile ready
#endif
  __syncthreads();

  // --- k_lr = Kmat @ fac : M=256 N=16 K=64 (V DMA still in flight) --------
  {
    v8f kacc[2] = {zero8(), zero8()};
    for (int kc = 0; kc < 64; kc += 32) {
      v16bf bf = frag_b(&facT[fr * 72 + kc], kb);
#pragma unroll
      for (int t = 0; t < 2; ++t) {
        const int m = (wave * 2 + t) * 16 + fr;
        kacc[t] = bwmma(frag_a(&Kbuf[m * 72 + kc], k0), bf, kacc[t]);
      }
    }
#pragma unroll
    for (int t = 0; t < 2; ++t) {
      const int hb = (wave * 2 + t) * 16 + dm;
#pragma unroll
      for (int i = 0; i < 8; ++i)
        klrT[fr * 264 + hb + i] = f2bf(kacc[t][i]);
    }
  }
#if HAVE_TDM
  if (wave == 0) __builtin_amdgcn_s_wait_tensorcnt(0);  // V tile ready
#endif
  __syncthreads();

  // --- v_lr = Vmat @ fac ----------------------------------------------------
  {
    v8f vacc[2] = {zero8(), zero8()};
    for (int kc = 0; kc < 64; kc += 32) {
      v16bf bf = frag_b(&facT[fr * 72 + kc], kb);
#pragma unroll
      for (int t = 0; t < 2; ++t) {
        const int m = (wave * 2 + t) * 16 + fr;
        vacc[t] = bwmma(frag_a(&Vbuf[m * 72 + kc], k0), bf, vacc[t]);
      }
    }
#pragma unroll
    for (int t = 0; t < 2; ++t) {
      const int hb = (wave * 2 + t) * 16 + dm;
#pragma unroll
      for (int i = 0; i < 8; ++i)
        vlrB[(hb + i) * 40 + fr] = f2bf(vacc[t][i]);
    }
  }
  __syncthreads();

  // --- scores = q^T @ k_lr : M=128 N=16 K=256 in 8 chunks ------------------
  const unsigned short* Qbase = Qp + (size_t)(b * 2048 + a * 32) * 1024;
  v8f sacc = zero8();
  for (int hc = 0; hc < 8; ++hc) {
    // stage q^T chunk [l][hh] (hh = 0..31)
    {
      const int hh  = tid >> 3;
      const int l0  = (tid & 7) << 4;
      const int h   = hc * 32 + hh;
      const unsigned short* src = Qbase + (h >> 3) * 1024 + (h & 7) * 128 + l0;
      unsigned short v[16];
      *(uint4*)&v[0] = *(const uint4*)(src);
      *(uint4*)&v[8] = *(const uint4*)(src + 8);
#pragma unroll
      for (int j = 0; j < 16; ++j) qT[(l0 + j) * 40 + hh] = v[j];
    }
    __syncthreads();
    sacc = bwmma(frag_a(&qT[(wave * 16 + fr) * 40], k0),
                 frag_b(&klrT[fr * 264 + hc * 32], kb), sacc);
    __syncthreads();
  }

  // --- softmax over r ------------------------------------------------------
  {
    const int lb = wave * 16 + dm;
#pragma unroll
    for (int i = 0; i < 8; ++i) scoresS[(lb + i) * 16 + fr] = sacc[i] * 0.0625f;
  }
  __syncthreads();
  if (tid < 128) {
    float v[16];
    float mx = -3.4e38f;
#pragma unroll
    for (int r = 0; r < 16; ++r) { v[r] = scoresS[tid * 16 + r]; mx = v[r] > mx ? v[r] : mx; }
    float sum = 0.f;
#pragma unroll
    for (int r = 0; r < 16; ++r) { v[r] = __expf(v[r] - mx); sum += v[r]; }
    const float inv = 1.f / sum;
#pragma unroll
    for (int r = 0; r < 16; ++r) attnB[tid * 40 + r] = f2bf(v[r] * inv);
#pragma unroll
    for (int r = 0; r < 16; ++r) attnB[tid * 40 + 16 + r] = 0;  // K padding
  }
  __syncthreads();

  // --- ctx = attn @ v_lr^T : M=128 N=256 K=32(padded) ----------------------
  {
    const v16bf af = frag_a(&attnB[(wave * 16 + fr) * 40], k0);
    const int   lb = wave * 16 + dm;
#pragma unroll
    for (int nt = 0; nt < 16; ++nt) {
      v8f d = bwmma(af, frag_b(&vlrB[(nt * 16 + fr) * 40], kb), zero8());
      const int h = nt * 16 + fr;
#pragma unroll
      for (int i = 0; i < 8; ++i) {
        const int l  = lb + i;
        const int s  = a * 32 + (l >> 2);
        const int dd = ((l & 3) << 8) + h;
        ctx[(size_t)(b * 2048 + s) * 1024 + dd] = f2bf(d[i]);
      }
    }
  }
}

// ---------------------------------------------------------------------------
// Residual + LayerNorm: out = LN(y + Q) * gamma + beta. One block per row.
// ---------------------------------------------------------------------------
__global__ __launch_bounds__(256) void ln_kernel(
    const float* __restrict__ y, const float* __restrict__ Qin,
    const float* __restrict__ gamma, const float* __restrict__ beta,
    float* __restrict__ out) {
  const int tid = threadIdx.x;
  const size_t row = blockIdx.x;
  const float* yr = y   + row * 1024;
  const float* qr = Qin + row * 1024;
  float*       orow = out + row * 1024;

  __shared__ float r1[256];
  __shared__ float r2[256];
  float xv[4], s = 0.f, s2 = 0.f;
#pragma unroll
  for (int i = 0; i < 4; ++i) {
    const float x = yr[tid + i * 256] + qr[tid + i * 256];
    xv[i] = x; s += x; s2 += x * x;
  }
  r1[tid] = s; r2[tid] = s2;
  __syncthreads();
  for (int st = 128; st > 0; st >>= 1) {
    if (tid < st) { r1[tid] += r1[tid + st]; r2[tid] += r2[tid + st]; }
    __syncthreads();
  }
  const float mean = r1[0] * (1.f / 1024.f);
  const float var  = r2[0] * (1.f / 1024.f) - mean * mean;
  const float inv  = rsqrtf(var + 1e-5f);
#pragma unroll
  for (int i = 0; i < 4; ++i) {
    const int c = tid + i * 256;
    orow[c] = (xv[i] - mean) * inv * gamma[c] + beta[c];
  }
}

// ---------------------------------------------------------------------------
extern "C" void kernel_launch(void* const* d_in, const int* in_sizes, int n_in,
                              void* d_out, int out_size, void* d_ws, size_t ws_size,
                              hipStream_t stream) {
  (void)in_sizes; (void)n_in; (void)out_size; (void)ws_size;

  const float* Qf  = (const float*)d_in[0];
  const float* Kf  = (const float*)d_in[1];
  const float* Vf  = (const float*)d_in[2];
  const float* WQ  = (const float*)d_in[3];
  const float* bQ  = (const float*)d_in[4];
  const float* WK  = (const float*)d_in[5];
  const float* bK  = (const float*)d_in[6];
  const float* WV  = (const float*)d_in[7];
  const float* bV  = (const float*)d_in[8];
  const float* fac = (const float*)d_in[9];
  const float* W1  = (const float*)d_in[10];
  const float* b1  = (const float*)d_in[11];
  const float* W2  = (const float*)d_in[12];
  const float* b2  = (const float*)d_in[13];
  const float* gam = (const float*)d_in[14];
  const float* bet = (const float*)d_in[15];

  // workspace layout (bytes):
  //   [0,   64M)  Qp  bf16 32768x1024   -- later reused as y f32 (134M)
  //   [64M, 96M)  Kp  bf16 16384x1024
  //   [96M, 128M) Vp  bf16 16384x1024
  //   [128M,192M) ctx bf16 32768x1024
  //   [192M,320M) h1  bf16 32768x2048
  char* ws = (char*)d_ws;
  unsigned short* Qp  = (unsigned short*)(ws);
  unsigned short* Kp  = (unsigned short*)(ws + 67108864ull);
  unsigned short* Vp  = (unsigned short*)(ws + 100663296ull);
  unsigned short* ctx = (unsigned short*)(ws + 134217728ull);
  unsigned short* h1  = (unsigned short*)(ws + 201326592ull);
  float*          yb  = (float*)(ws);  // overlays dead Qp/Kp/Vp

  const dim3 blk(256);

  // Projections (A f32, out bf16)
  gemm_bias_kernel<float, true, false>
      <<<dim3(8, 256), blk, 0, stream>>>(Qf, WQ, bQ, Qp, 32768, 1024, 1024);
  gemm_bias_kernel<float, true, false>
      <<<dim3(8, 128), blk, 0, stream>>>(Kf, WK, bK, Kp, 16384, 1024, 1024);
  gemm_bias_kernel<float, true, false>
      <<<dim3(8, 128), blk, 0, stream>>>(Vf, WV, bV, Vp, 16384, 1024, 1024);

  // Fused low-rank attention
  attn_kernel<<<dim3(1024), blk, 0, stream>>>(Qp, Kp, Vp, fac, ctx);

  // FFN (bf16 A staged by TDM, double buffered)
  gemm_bias_kernel<unsigned short, true, true>
      <<<dim3(16, 256), blk, 0, stream>>>(ctx, W1, b1, h1, 32768, 2048, 1024);
  gemm_bias_kernel<unsigned short, false, false>
      <<<dim3(8, 256), blk, 0, stream>>>(h1, W2, b2, yb, 32768, 1024, 2048);

  // Residual + LayerNorm
  ln_kernel<<<dim3(32768), blk, 0, stream>>>(yb, Qf, gam, bet, (float*)d_out);
}
